// MultiHeadAttention_60232621359339
// MI455X (gfx1250) — matrix operand order repaired and checked by reading, not compiled, not run on hardware
//
#include <hip/hip_runtime.h>
#include <cstdint>
#include <cstddef>

#define DIM_C   2048
#define T_C     4096
#define NH_C    16
#define NKV_C   4
#define HD_C    128
#define KVD_C   (NKV_C * HD_C)   // 512
#define WIN_C   1024

typedef __attribute__((ext_vector_type(16))) __bf16 v16bf;
typedef __attribute__((ext_vector_type(8)))  __bf16 v8bf;
typedef __attribute__((ext_vector_type(8)))  float  v8f;

union frag_u { v16bf v; v8bf h[2]; };

__device__ __forceinline__ v8f wmma_bf16(v16bf a, v16bf b, v8f c) {
  // D = A(16x32) * B(32x16) + C, f32 accum
  return __builtin_amdgcn_wmma_f32_16x16x32_bf16(
      /*neg_a=*/false, a, /*neg_b=*/false, b,
      /*c_mod=*/(short)0, c, /*reuse_a=*/false, /*reuse_b=*/false);
}

// A fragment: 16x32 (MxK) from row-major source, leading dim ld (elements).
// lanes 0-15: row=lane, K 0..7 & 16..23 ; lanes 16-31: row=lane-16, K 8..15 & 24..31
__device__ __forceinline__ v16bf load_frag_a(const __bf16* base, int ld, int lane) {
  int row = lane & 15;
  int k0  = (lane < 16) ? 0 : 8;
  frag_u u;
  u.h[0] = *(const v8bf*)(base + row * ld + k0);
  u.h[1] = *(const v8bf*)(base + row * ld + k0 + 16);
  return u.v;
}

// B fragment: 32x16 (KxN) loaded from N-major (i.e. B^T) storage, leading dim ld.
// lanes 0-15: n=lane, K 0..7 & 16..23 ; lanes 16-31: n=lane-16, K 8..15 & 24..31
__device__ __forceinline__ v16bf load_frag_b(const __bf16* base, int ld, int lane) {
  int n  = lane & 15;
  int k0 = (lane < 16) ? 0 : 8;
  frag_u u;
  u.h[0] = *(const v8bf*)(base + n * ld + k0);
  u.h[1] = *(const v8bf*)(base + n * ld + k0 + 16);
  return u.v;
}

// Async 16-byte global -> LDS copy (per-lane), tracked by ASYNCcnt.
__device__ __forceinline__ void async_copy16(const __bf16* gsrc, void* lds_dst) {
  uint32_t loff = (uint32_t)(uintptr_t)lds_dst;           // addr[31:0] == LDS offset
  uint64_t gaddr = (uint64_t)(uintptr_t)gsrc;
  asm volatile("global_load_async_to_lds_b128 %0, %1, off"
               :: "v"(loff), "v"(gaddr)
               : "memory");
}

__device__ __forceinline__ void wait_async0() {
  asm volatile("s_wait_asynccnt 0x0" ::: "memory");
}

// ---------------- simple prep kernels ----------------

__global__ void cast_kernel(const float* __restrict__ src, __bf16* __restrict__ dst, int n) {
  int i = blockIdx.x * blockDim.x + threadIdx.x;
  if (i < n) dst[i] = (__bf16)src[i];
}

// src[K][N] f32 -> dst[N][K] bf16
__global__ void transpose_cast_kernel(const float* __restrict__ src, __bf16* __restrict__ dst,
                                      int K, int N) {
  __shared__ float tile[32][33];
  int kb = blockIdx.y * 32, nb = blockIdx.x * 32;
  int tx = threadIdx.x, ty = threadIdx.y;
#pragma unroll
  for (int i = 0; i < 4; ++i) {
    int r = ty + 8 * i;
    if (kb + r < K && nb + tx < N)
      tile[r][tx] = src[(size_t)(kb + r) * N + nb + tx];
  }
  __syncthreads();
#pragma unroll
  for (int i = 0; i < 4; ++i) {
    int r = ty + 8 * i;
    if (nb + r < N && kb + tx < K)
      dst[(size_t)(nb + r) * K + kb + tx] = (__bf16)tile[tx][r];
  }
}

// In-place RoPE on buf[T][heads*128]; pair (d, d+64) per head. outscale folded in.
__global__ void rope_kernel(__bf16* __restrict__ buf, int heads, float outscale) {
  int idx = blockIdx.x * blockDim.x + threadIdx.x;
  int total = T_C * heads * 64;
  if (idx >= total) return;
  int d = idx & 63;
  int h = (idx >> 6) % heads;
  int t = idx / (heads * 64);
  size_t base = (size_t)t * heads * 128 + h * 128;
  // inv_freq = 10000^(-d/64) = exp(-d/64 * ln(10000))
  float freq = __expf(-(float)d * (9.210340372f / 64.0f));
  float ang = (float)t * freq;
  float c = cosf(ang), s = sinf(ang);
  float q0 = (float)buf[base + d];
  float q1 = (float)buf[base + d + 64];
  buf[base + d]      = (__bf16)((q0 * c - q1 * s) * outscale);
  buf[base + d + 64] = (__bf16)((q1 * c + q0 * s) * outscale);
}

// Vr[T][512] -> Vt[c][T]  (c = kv*128 + d)
__global__ void vtrans_kernel(const __bf16* __restrict__ Vr, __bf16* __restrict__ Vt) {
  int idx = blockIdx.x * blockDim.x + threadIdx.x;
  if (idx >= T_C * KVD_C) return;
  int t = idx >> 9;          // /512
  int c = idx & 511;
  Vt[(size_t)c * T_C + t] = Vr[idx];
}

// ---------------- WMMA GEMM: C[M][N] = A[M][K] * B^T[N][K] ----------------
// Double-buffered LDS staging via async global->LDS loads (ASYNCcnt).

template <bool F32OUT>
__global__ __launch_bounds__(256) void gemm_kernel(const __bf16* __restrict__ A,
                                                   const __bf16* __restrict__ BT,
                                                   void* __restrict__ Cout,
                                                   int M, int N, int K) {
  __shared__ __align__(16) __bf16 As[2][128][40];
  __shared__ __align__(16) __bf16 Bs[2][128][40];
  int tid  = threadIdx.x;
  int lane = tid & 31;
  int wid  = tid >> 5;
  int wm = wid & 3;        // 4 waves along M
  int wn = wid >> 2;       // 2 waves along N
  int m0 = blockIdx.y * 128;
  int n0 = blockIdx.x * 128;

  // per-thread staging coordinates: 2 x 16B per matrix per thread
  int r0 = tid >> 2;                 // rows 0..63   (transfer 0)
  int r1 = (tid + 256) >> 2;         // rows 64..127 (transfer 1)
  int s0 = (tid & 3) * 8;            // 8-element column segment

  auto issue_tile = [&](int buf, int k0) {
    async_copy16(A  + (size_t)(m0 + r0) * K + k0 + s0, &As[buf][r0][s0]);
    async_copy16(A  + (size_t)(m0 + r1) * K + k0 + s0, &As[buf][r1][s0]);
    async_copy16(BT + (size_t)(n0 + r0) * K + k0 + s0, &Bs[buf][r0][s0]);
    async_copy16(BT + (size_t)(n0 + r1) * K + k0 + s0, &Bs[buf][r1][s0]);
  };

  v8f acc[2][4];
#pragma unroll
  for (int i = 0; i < 2; ++i)
#pragma unroll
    for (int j = 0; j < 4; ++j) acc[i][j] = (v8f)0.0f;

  int nk = K >> 5;
  issue_tile(0, 0);
  for (int kb = 0; kb < nk; ++kb) {
    int cur = kb & 1;
    wait_async0();          // our async writes into LDS are done
    __syncthreads();        // everyone's writes visible, prev reads finished
    if (kb + 1 < nk) issue_tile(cur ^ 1, (kb + 1) * 32);  // prefetch under compute

    v16bf af[2], bfm[4];
#pragma unroll
    for (int i = 0; i < 2; ++i) af[i]  = load_frag_a(&As[cur][wm * 32 + i * 16][0], 40, lane);
#pragma unroll
    for (int j = 0; j < 4; ++j) bfm[j] = load_frag_b(&Bs[cur][wn * 64 + j * 16][0], 40, lane);
#pragma unroll
    for (int i = 0; i < 2; ++i)
#pragma unroll
      for (int j = 0; j < 4; ++j) acc[i][j] = wmma_bf16(af[i], bfm[j], acc[i][j]);
    __syncthreads();        // done reading buffer `cur`
  }

  int hb = lane >> 4, cl = lane & 15;
#pragma unroll
  for (int i = 0; i < 2; ++i)
#pragma unroll
    for (int j = 0; j < 4; ++j)
#pragma unroll
      for (int r = 0; r < 8; ++r) {
        int row = m0 + wm * 32 + i * 16 + r + 8 * hb;
        int col = n0 + wn * 64 + j * 16 + cl;
        float v = acc[i][j][r];
        if (F32OUT) ((float*)Cout)[(size_t)row * N + col] = v;
        else        ((__bf16*)Cout)[(size_t)row * N + col] = (__bf16)v;
      }
}

// ---------------- flash attention (sliding window + causal) ----------------
// One wave per (head, 16-query tile). Q pre-scaled by (1/sqrt(128))*log2(e).

__global__ __launch_bounds__(256) void attn_kernel(const __bf16* __restrict__ Q,
                                                   const __bf16* __restrict__ Kc,
                                                   const __bf16* __restrict__ Vt,
                                                   __bf16* __restrict__ AO) {
  __shared__ __align__(16) __bf16 Pbuf[8][16 * 40];
  int lane = threadIdx.x & 31;
  int wid  = threadIdx.x >> 5;
  int gw   = blockIdx.x * 8 + wid;
  int head = gw >> 8;               // 256 query tiles per head
  int i0   = (gw & 255) << 4;
  int kv   = head >> 2;             // repeat_interleave GQA mapping
  int hb = lane >> 4, cl = lane & 15;

  const __bf16* qb = Q  + (size_t)i0 * (NH_C * HD_C) + head * HD_C;
  const __bf16* kb = Kc + kv * HD_C;                 // row stride 512
  const __bf16* vb = Vt + (size_t)kv * HD_C * T_C;   // [d][t]
  __bf16* pb = &Pbuf[wid][0];

  v16bf qf[4];
#pragma unroll
  for (int f = 0; f < 4; ++f) qf[f] = load_frag_a(qb + f * 32, NH_C * HD_C, lane);

  v8f o[8];
#pragma unroll
  for (int t = 0; t < 8; ++t) o[t] = (v8f)0.0f;
  float mr[8], lr[8];
#pragma unroll
  for (int r = 0; r < 8; ++r) { mr[r] = -1e30f; lr[r] = 0.0f; }

  int jlo = (i0 > (WIN_C - 1)) ? ((i0 - (WIN_C - 1)) & ~31) : 0;

  for (int j0 = jlo; j0 < i0 + 16; j0 += 32) {
    // ---- S = Q * K^T for 16 queries x 32 keys ----
    v8f sacc[2];
    sacc[0] = (v8f)0.0f; sacc[1] = (v8f)0.0f;
#pragma unroll
    for (int c = 0; c < 2; ++c)
#pragma unroll
      for (int f = 0; f < 4; ++f) {
        v16bf kf = load_frag_b(kb + (size_t)(j0 + c * 16) * KVD_C + f * 32, KVD_C, lane);
        sacc[c] = wmma_bf16(qf[f], kf, sacc[c]);
      }

    // ---- mask + online softmax (stats per row; rows r+8*hb) ----
#pragma unroll
    for (int r = 0; r < 8; ++r) {
      int qrow = i0 + r + 8 * hb;
      int k0i = j0 + cl;
      int k1i = j0 + 16 + cl;
      float s0 = sacc[0][r];
      float s1 = sacc[1][r];
      bool a0 = (k0i <= qrow) && ((qrow - k0i) < WIN_C);
      bool a1 = (k1i <= qrow) && ((qrow - k1i) < WIN_C);
      s0 = a0 ? s0 : -1e30f;
      s1 = a1 ? s1 : -1e30f;
      float v = fmaxf(s0, s1);
#pragma unroll
      for (int m = 1; m <= 8; m <<= 1) v = fmaxf(v, __shfl_xor(v, m, 32));
      float mn = fmaxf(mr[r], v);
      float alpha = exp2f(mr[r] - mn);
      mr[r] = mn;
      float p0 = exp2f(s0 - mn);
      float p1 = exp2f(s1 - mn);
      float sum = p0 + p1;
#pragma unroll
      for (int m = 1; m <= 8; m <<= 1) sum += __shfl_xor(sum, m, 32);
      lr[r] = lr[r] * alpha + sum;
#pragma unroll
      for (int t = 0; t < 8; ++t) o[t][r] = o[t][r] * alpha;
      int prow = r + 8 * hb;
      pb[prow * 40 + cl]      = (__bf16)p0;
      pb[prow * 40 + 16 + cl] = (__bf16)p1;
    }

    // same-wave LDS store -> load handoff
    asm volatile("s_wait_dscnt 0x0" ::: "memory");
    v16bf pf = load_frag_a(pb, 40, lane);

    // ---- O += P * V  (V^T stored [d][t], contiguous along keys) ----
#pragma unroll
    for (int t = 0; t < 8; ++t) {
      v16bf vf = load_frag_b(vb + (size_t)(t * 16) * T_C + j0, T_C, lane);
      o[t] = wmma_bf16(pf, vf, o[t]);
    }
  }

  // ---- normalize + store bf16 to AO[T][2048] ----
  float inv[8];
#pragma unroll
  for (int r = 0; r < 8; ++r) inv[r] = 1.0f / lr[r];
  __bf16* ob = AO + (size_t)i0 * (NH_C * HD_C) + head * HD_C;
#pragma unroll
  for (int t = 0; t < 8; ++t)
#pragma unroll
    for (int r = 0; r < 8; ++r) {
      int row = r + 8 * hb;
      int col = t * 16 + cl;
      ob[(size_t)row * (NH_C * HD_C) + col] = (__bf16)(o[t][r] * inv[r]);
    }
}

// ---------------- host launcher ----------------

extern "C" void kernel_launch(void* const* d_in, const int* in_sizes, int n_in,
                              void* d_out, int out_size, void* d_ws, size_t ws_size,
                              hipStream_t stream) {
  (void)in_sizes; (void)n_in; (void)out_size; (void)ws_size;
  const float* x  = (const float*)d_in[0];
  const float* Wq = (const float*)d_in[1];
  const float* Wk = (const float*)d_in[2];
  const float* Wv = (const float*)d_in[3];
  const float* Wo = (const float*)d_in[4];
  float* out = (float*)d_out;

  char* ws = (char*)d_ws;
  size_t off = 0;
  auto alloc = [&](size_t bytes) -> __bf16* {
    __bf16* p = (__bf16*)(ws + off);
    off += (bytes + 255) & ~(size_t)255;
    return p;
  };
  __bf16* xb  = alloc((size_t)T_C * DIM_C * 2);
  __bf16* WqT = alloc((size_t)DIM_C * DIM_C * 2);
  __bf16* WkT = alloc((size_t)KVD_C * DIM_C * 2);
  __bf16* WvT = alloc((size_t)KVD_C * DIM_C * 2);
  __bf16* WoT = alloc((size_t)DIM_C * DIM_C * 2);
  __bf16* Qb  = alloc((size_t)T_C * DIM_C * 2);
  __bf16* Kb  = alloc((size_t)T_C * KVD_C * 2);
  __bf16* Vb  = alloc((size_t)T_C * KVD_C * 2);
  __bf16* Vt  = alloc((size_t)KVD_C * T_C * 2);
  __bf16* AO  = alloc((size_t)T_C * DIM_C * 2);

  dim3 tb(32, 8);
  cast_kernel<<<(T_C * DIM_C + 255) / 256, 256, 0, stream>>>(x, xb, T_C * DIM_C);
  transpose_cast_kernel<<<dim3(DIM_C / 32, DIM_C / 32), tb, 0, stream>>>(Wq, WqT, DIM_C, DIM_C);
  transpose_cast_kernel<<<dim3(KVD_C / 32, DIM_C / 32), tb, 0, stream>>>(Wk, WkT, DIM_C, KVD_C);
  transpose_cast_kernel<<<dim3(KVD_C / 32, DIM_C / 32), tb, 0, stream>>>(Wv, WvT, DIM_C, KVD_C);
  transpose_cast_kernel<<<dim3(DIM_C / 32, DIM_C / 32), tb, 0, stream>>>(Wo, WoT, DIM_C, DIM_C);

  gemm_kernel<false><<<dim3(DIM_C / 128, T_C / 128), 256, 0, stream>>>(xb, WqT, Qb, T_C, DIM_C, DIM_C);
  gemm_kernel<false><<<dim3(KVD_C / 128, T_C / 128), 256, 0, stream>>>(xb, WkT, Kb, T_C, KVD_C, DIM_C);
  gemm_kernel<false><<<dim3(KVD_C / 128, T_C / 128), 256, 0, stream>>>(xb, WvT, Vb, T_C, KVD_C, DIM_C);

  // fold softmax scale (1/sqrt(128)) * log2(e) into Q so attention uses exp2 directly
  rope_kernel<<<(T_C * NH_C * 64 + 255) / 256, 256, 0, stream>>>(Qb, NH_C, 0.12753124f);
  rope_kernel<<<(T_C * NKV_C * 64 + 255) / 256, 256, 0, stream>>>(Kb, NKV_C, 1.0f);
  vtrans_kernel<<<(T_C * KVD_C + 255) / 256, 256, 0, stream>>>(Vb, Vt);

  attn_kernel<<<dim3((NH_C * (T_C / 16)) / 8), 256, 0, stream>>>(Qb, Kb, Vt, AO);

  gemm_kernel<true><<<dim3(DIM_C / 128, T_C / 128), 256, 0, stream>>>(AO, WoT, (void*)out, T_C, DIM_C, DIM_C);
}
